// PipelineAlign_8993661518642
// MI455X (gfx1250) — compile-verified, matrix operand order checked by reading, not run on hardware
//
#include <hip/hip_runtime.h>
#include <hip/hip_bf16.h>
#include <stdint.h>

// ---------------------------------------------------------------------------
// Problem constants (match the JAX reference)
// ---------------------------------------------------------------------------
#define SRC_SIZE 10000
#define TRG_SIZE 10000
#define EMBED    512
#define HID      512
#define GATES    (4 * HID)       // 2048
#define BB       64
#define SS       256
#define TT       256
#define ROWS     (BB * SS)       // 16384
#define OUTH     (2 * HID)       // 1024

typedef __bf16 bf16_t;
typedef __attribute__((ext_vector_type(16))) __bf16 v16bf;
typedef __attribute__((ext_vector_type(8)))  float  v8f;

// ---------------------------------------------------------------------------
// Utility kernels
// ---------------------------------------------------------------------------
__global__ void zero_f32(float* __restrict__ p, size_t n) {
    size_t i = (size_t)blockIdx.x * blockDim.x + threadIdx.x;
    size_t stride = (size_t)gridDim.x * blockDim.x;
    for (; i < n; i += stride) p[i] = 0.0f;
}

__global__ void cast_f32_bf16(const float* __restrict__ src, bf16_t* __restrict__ dst, size_t n) {
    size_t i = (size_t)blockIdx.x * blockDim.x + threadIdx.x;
    size_t stride = (size_t)gridDim.x * blockDim.x;
    for (; i < n; i += stride) dst[i] = (bf16_t)src[i];
}

// Gather embedding rows by token id and cast fp32 -> bf16.
__global__ void gather_cast(const int* __restrict__ tok, const float* __restrict__ emb,
                            bf16_t* __restrict__ dst) {
    size_t n = (size_t)ROWS * EMBED;
    size_t i = (size_t)blockIdx.x * blockDim.x + threadIdx.x;
    size_t stride = (size_t)gridDim.x * blockDim.x;
    for (; i < n; i += stride) {
        size_t m = i / EMBED;
        size_t e = i - m * EMBED;
        dst[i] = (bf16_t)emb[(size_t)tok[m] * EMBED + e];
    }
}

__global__ void zero_state(float* __restrict__ c, bf16_t* __restrict__ h) {
    int i = blockIdx.x * blockDim.x + threadIdx.x;
    if (i < BB * HID) { c[i] = 0.0f; h[i] = (bf16_t)0.0f; }
}

// ---------------------------------------------------------------------------
// bf16 WMMA GEMM with async-to-LDS weight staging.
//   D[M,N] = A[M,K] * Bw[N,K]^T (+ addsrc) (+ bias)
// Block: 128 threads (4 waves), block tile 64(M) x 64(N).
// Each wave computes a 16x64 strip: 4 f32 accumulators, one shared A fragment.
// Bw tile (64 rows x 32 K, 4 KB) is double-buffered in LDS and filled with
// GLOBAL_LOAD_ASYNC_TO_LDS_B128 (ASYNCcnt), prefetching chunk c+1 during the
// WMMAs of chunk c; fenced with s_wait_asynccnt 0 + workgroup barrier.
// ---------------------------------------------------------------------------
__global__ void __launch_bounds__(128)
gemm_bf16_lds(const bf16_t* __restrict__ A,
              const bf16_t* __restrict__ Bw,
              const float* __restrict__ bias,
              const float* __restrict__ addsrc, int ld_add,
              float* __restrict__ D, int ldd,
              int M, int N, int K) {
    __shared__ bf16_t tileB[2][64 * 32];          // 2 x 4 KB weight tiles

    const int tid  = threadIdx.x;
    const int lane = tid & 31;
    const int wave = tid >> 5;
    const int l  = lane & 15;
    const int hi = lane >> 4;

    const int m0   = blockIdx.y * 64 + wave * 16;  // wave-private M strip
    const int nblk = blockIdx.x * 64;              // block N tile

    // --- accumulator init: D element r -> (m = r + hi*8, n = l) ---
    v8f acc[4];
#pragma unroll
    for (int j = 0; j < 4; ++j) {
#pragma unroll
        for (int r = 0; r < 8; ++r) {
            int m = r + hi * 8;
            int n = nblk + j * 16 + l;
            float c0 = 0.0f;
            if (addsrc) c0 = addsrc[(size_t)(m0 + m) * ld_add + n];
            if (bias)   c0 += bias[n];
            acc[j][r] = c0;
        }
    }

    // Low 32 bits of the generic shared pointer == LDS byte offset
    // (flat LDS aperture).
    const unsigned ldsBase = (unsigned)(uintptr_t)&tileB[0][0];

    // Issue one 32-K chunk of the 64x32 B tile: 256 x 16B segments,
    // 2 async b128 per thread (ASYNCcnt += 2 per wave-issue pair).
    auto issue_chunk = [&](int kc, int buf) {
#pragma unroll
        for (int s = 0; s < 2; ++s) {
            int seg = tid + s * 128;               // 0..255
            int row = seg >> 2;                    // 0..63
            int kp  = seg & 3;                     // 16B piece within row
            const bf16_t* src = Bw + (size_t)(nblk + row) * K + kc + kp * 8;
            unsigned dst = ldsBase + (unsigned)buf * 4096u + (unsigned)seg * 16u;
            asm volatile("global_load_async_to_lds_b128 %0, %1, off"
                         :: "v"(dst), "v"((unsigned long long)(uintptr_t)src)
                         : "memory");
        }
    };

    issue_chunk(0, 0);
    asm volatile("s_wait_asynccnt 0" ::: "memory");
    __syncthreads();

    const bf16_t* arow = A + (size_t)(m0 + l) * K;
    const int nchunks = K >> 5;

    for (int c = 0; c < nchunks; ++c) {
        const int cur = c & 1;
        if (c + 1 < nchunks) issue_chunk((c + 1) << 5, cur ^ 1);

        // A fragment (16x32 bf16) straight from global:
        // VGPR v holds K = (v>>2)*16 + hi*8 + (v&3)*2 (+0,+1)
        v16bf a;
#pragma unroll
        for (int v = 0; v < 8; ++v) {
            int k = (c << 5) + ((v >> 2) << 4) + (hi << 3) + ((v & 3) << 1);
            a[2 * v]     = arow[k];
            a[2 * v + 1] = arow[k + 1];
        }

        // 4 B fragments from LDS (two contiguous 16B rows each -> ds_load_b128)
#pragma unroll
        for (int j = 0; j < 4; ++j) {
            const bf16_t* brow = &tileB[cur][(j * 16 + l) * 32];
            v16bf b;
#pragma unroll
            for (int v = 0; v < 8; ++v) {
                int k = ((v >> 2) << 4) + (hi << 3) + ((v & 3) << 1);
                b[2 * v]     = brow[k];
                b[2 * v + 1] = brow[k + 1];
            }
            acc[j] = __builtin_amdgcn_wmma_f32_16x16x32_bf16(
                false, a, false, b, (short)0, acc[j], false, false);
        }

        asm volatile("s_wait_asynccnt 0" ::: "memory");
        __syncthreads();
    }

#pragma unroll
    for (int j = 0; j < 4; ++j) {
#pragma unroll
        for (int r = 0; r < 8; ++r) {
            int m = r + hi * 8;
            D[(size_t)(m0 + m) * ldd + nblk + j * 16 + l] = acc[j][r];
        }
    }
}

// ---------------------------------------------------------------------------
// LSTM gate update for one timestep.
// ---------------------------------------------------------------------------
__device__ __forceinline__ float sigm(float x) { return 1.0f / (1.0f + __expf(-x)); }

__global__ void lstm_gate(const float* __restrict__ g,
                          float* __restrict__ c_st, bf16_t* __restrict__ h_st,
                          bf16_t* __restrict__ out, int t_eff, int dir) {
    int i = blockIdx.x * blockDim.x + threadIdx.x;
    if (i >= BB * HID) return;
    int b = i / HID;
    int h = i - b * HID;
    const float* gr = g + (size_t)b * GATES;
    float i_ = sigm(gr[h]);
    float f_ = sigm(gr[HID + h]);
    float g_ = tanhf(gr[2 * HID + h]);
    float o_ = sigm(gr[3 * HID + h]);
    float c  = f_ * c_st[i] + i_ * g_;
    float hh = o_ * tanhf(c);
    c_st[i] = c;
    h_st[i] = (bf16_t)hh;
    out[(size_t)b * SS * OUTH + (size_t)t_eff * OUTH + (size_t)dir * HID + h] = (bf16_t)hh;
}

// ---------------------------------------------------------------------------
// Cross similarity + masked scatter-add (fused epilogue; logits stay in VGPRs).
// sigmoid(x) > 0.5  <=>  x > 0
// ---------------------------------------------------------------------------
__global__ void cross_scatter(const bf16_t* __restrict__ src_out,
                              const bf16_t* __restrict__ trg_out,
                              const int* __restrict__ s_tok,
                              const int* __restrict__ t_tok,
                              float* __restrict__ lex) {
    const int lane = threadIdx.x & 31;
    const int wave = threadIdx.x >> 5;
    const int l  = lane & 15;
    const int hi = lane >> 4;
    const int b  = blockIdx.z;
    const int m0 = blockIdx.y * 16;
    const int n0 = (blockIdx.x * 4 + wave) * 16;

    const bf16_t* A  = src_out + (size_t)b * SS * OUTH;
    const bf16_t* Bm = trg_out + (size_t)b * SS * OUTH;

    v8f acc = {};
    const bf16_t* arow = A  + (size_t)(m0 + l) * OUTH;
    const bf16_t* brow = Bm + (size_t)(n0 + l) * OUTH;
    for (int k0 = 0; k0 < OUTH; k0 += 32) {
        v16bf a, bfrag;
#pragma unroll
        for (int v = 0; v < 8; ++v) {
            int k = k0 + ((v >> 2) << 4) + (hi << 3) + ((v & 3) << 1);
            a[2 * v]         = arow[k];
            a[2 * v + 1]     = arow[k + 1];
            bfrag[2 * v]     = brow[k];
            bfrag[2 * v + 1] = brow[k + 1];
        }
        acc = __builtin_amdgcn_wmma_f32_16x16x32_bf16(
            false, a, false, bfrag, (short)0, acc, false, false);
    }

    const int* srow = s_tok + (size_t)b * SS;
    const int* trow = t_tok + (size_t)b * TT;
    const int tcol = trow[n0 + l];
#pragma unroll
    for (int r = 0; r < 8; ++r) {
        int m = r + hi * 8;
        if (acc[r] > 0.0f) {
            size_t idx = (size_t)srow[m0 + m] * TRG_SIZE + (size_t)tcol;
            atomicAdd(&lex[idx], 1.0f);
        }
    }
}

// ---------------------------------------------------------------------------
// Host orchestration
// ---------------------------------------------------------------------------
extern "C" void kernel_launch(void* const* d_in, const int* in_sizes, int n_in,
                              void* d_out, int out_size, void* d_ws, size_t ws_size,
                              hipStream_t stream) {
    (void)in_sizes; (void)n_in; (void)ws_size;

    const int*   s_tok   = (const int*)d_in[0];
    const int*   t_tok   = (const int*)d_in[1];
    const float* src_emb = (const float*)d_in[2];
    const float* trg_emb = (const float*)d_in[3];
    const float* Wih[2][2]  = {{(const float*)d_in[4],  (const float*)d_in[7]},
                               {(const float*)d_in[10], (const float*)d_in[13]}};
    const float* Whh[2][2]  = {{(const float*)d_in[5],  (const float*)d_in[8]},
                               {(const float*)d_in[11], (const float*)d_in[14]}};
    const float* bias[2][2] = {{(const float*)d_in[6],  (const float*)d_in[9]},
                               {(const float*)d_in[12], (const float*)d_in[15]}};

    char* base = (char*)d_ws;
    size_t off = 0;
    auto carve = [&](size_t bytes) -> void* {
        off = (off + 255) & ~(size_t)255;
        void* p = base + off;
        off += bytes;
        return p;
    };
    float*  xw     = (float*) carve((size_t)ROWS * GATES * sizeof(float));
    float*  gbuf   = (float*) carve((size_t)BB * GATES * sizeof(float));
    bf16_t* hstate = (bf16_t*)carve((size_t)BB * HID * sizeof(bf16_t));
    float*  cstate = (float*) carve((size_t)BB * HID * sizeof(float));
    bf16_t* x_bf[2];
    x_bf[0] = (bf16_t*)carve((size_t)ROWS * EMBED * sizeof(bf16_t));
    x_bf[1] = (bf16_t*)carve((size_t)ROWS * EMBED * sizeof(bf16_t));
    bf16_t* wih_bf[2][2];
    bf16_t* whh_bf[2][2];
    for (int p = 0; p < 2; ++p)
        for (int d = 0; d < 2; ++d) {
            wih_bf[p][d] = (bf16_t*)carve((size_t)GATES * EMBED * sizeof(bf16_t));
            whh_bf[p][d] = (bf16_t*)carve((size_t)GATES * HID * sizeof(bf16_t));
        }
    bf16_t* out_bf[2];
    out_bf[0] = (bf16_t*)carve((size_t)ROWS * OUTH * sizeof(bf16_t));
    out_bf[1] = (bf16_t*)carve((size_t)ROWS * OUTH * sizeof(bf16_t));

    // 1. zero the lexicon (400 MB output)
    zero_f32<<<8192, 256, 0, stream>>>((float*)d_out, (size_t)out_size);

    // 2. gather + cast embeddings
    gather_cast<<<8192, 256, 0, stream>>>(s_tok, src_emb, x_bf[0]);
    gather_cast<<<8192, 256, 0, stream>>>(t_tok, trg_emb, x_bf[1]);

    // 3. cast weights to bf16
    for (int p = 0; p < 2; ++p)
        for (int d = 0; d < 2; ++d) {
            cast_f32_bf16<<<2048, 256, 0, stream>>>(Wih[p][d], wih_bf[p][d],
                                                    (size_t)GATES * EMBED);
            cast_f32_bf16<<<2048, 256, 0, stream>>>(Whh[p][d], whh_bf[p][d],
                                                    (size_t)GATES * HID);
        }

    // 4. BiLSTM for both sides
    for (int p = 0; p < 2; ++p) {
        for (int d = 0; d < 2; ++d) {
            // 4a. input projection: xw = x @ Wih^T + b   (M=16384, N=2048, K=512)
            {
                dim3 grid(GATES / 64, ROWS / 64);
                gemm_bf16_lds<<<grid, 128, 0, stream>>>(
                    x_bf[p], wih_bf[p][d], bias[p][d],
                    nullptr, 0, xw, GATES, ROWS, GATES, EMBED);
            }
            // 4b. recurrence (256 sequential steps)
            zero_state<<<(BB * HID + 255) / 256, 256, 0, stream>>>(cstate, hstate);
            for (int step = 0; step < SS; ++step) {
                int t_eff = (d == 0) ? step : (SS - 1 - step);
                dim3 grid(GATES / 64, BB / 64);   // (32, 1)
                gemm_bf16_lds<<<grid, 128, 0, stream>>>(
                    hstate, whh_bf[p][d], nullptr,
                    xw + (size_t)t_eff * GATES, SS * GATES,
                    gbuf, GATES, BB, GATES, HID);
                lstm_gate<<<(BB * HID + 255) / 256, 256, 0, stream>>>(
                    gbuf, cstate, hstate, out_bf[p], t_eff, d);
            }
        }
    }

    // 5. cross similarity + fused masked scatter-add
    {
        dim3 grid(TT / 64, SS / 16, BB);
        cross_scatter<<<grid, 128, 0, stream>>>(out_bf[0], out_bf[1],
                                                s_tok, t_tok, (float*)d_out);
    }
}
